// MSDeformAttn_7095285973221
// MI455X (gfx1250) — matrix operand order repaired and checked by reading, not compiled, not run on hardware
//
#include <hip/hip_runtime.h>
#include <hip/hip_bf16.h>

// ---- static problem config (matches reference) ----
#define NB     4
#define LQN    5440
#define LEN_IN 5440
#define DM     256
#define MH     8
#define DH     32
#define NQ_TOT (NB * LQN)          // 21760

typedef _Float16 h8  __attribute__((ext_vector_type(8)));
typedef _Float16 v16h __attribute__((ext_vector_type(16)));
typedef float    v8f  __attribute__((ext_vector_type(8)));

// ---------------------------------------------------------------------------
// (Mtot x 256) * (256 x NOUT) + bias GEMM, fp32 in/out, f16 WMMA core.
// Block = 256 threads (8 waves). Block tile = 64 rows x 64 cols, full K=256
// staged in LDS as f16 (A: 32KB, B^T: 32KB). Each wave owns two 16x16 C tiles
// sharing the same A fragment (2x A reuse): 16 wmma / wave, 3 ds_load_b128
// per wmma. NOUT is a template constant so the epilogue is base + immediate
// offsets (no u64 mults, no exec branches). Mtot must be a multiple of 64.
// ---------------------------------------------------------------------------
template <int NOUT>
__global__ __launch_bounds__(256) void gemm256_wmma(
    const float* __restrict__ A, const float* __restrict__ B,
    const float* __restrict__ bias, float* __restrict__ C)
{
    __shared__ __align__(32) _Float16 As[64 * 256];   // 32 KB
    __shared__ __align__(32) _Float16 Bs[64 * 256];   // 32 KB (B transposed: [n][k])

    const int tid  = threadIdx.x;
    const int row0 = blockIdx.x * 64;
    const int col0 = blockIdx.y * 64;

    // --- stage A tile (64 x 256), f32 -> f16, 64 elems / thread ---
    {
        const int r  = tid >> 2;          // 0..63
        const int kb = (tid & 3) * 64;    // 0,64,128,192
        const float4* src = (const float4*)(A + (size_t)(row0 + r) * 256 + kb);
#pragma unroll
        for (int i = 0; i < 16; ++i) {
            float4 v = src[i];
            int o = r * 256 + kb + i * 4;
            As[o + 0] = (_Float16)v.x; As[o + 1] = (_Float16)v.y;
            As[o + 2] = (_Float16)v.z; As[o + 3] = (_Float16)v.w;
        }
    }
    // --- stage B tile transposed (64 cols x 256 K), 64 elems / thread ---
    {
        const int nn = tid & 63;
        const int ks = (tid >> 6) * 64;
#pragma unroll 8
        for (int k = 0; k < 64; ++k)
            Bs[nn * 256 + ks + k] = (_Float16)B[(size_t)(ks + k) * NOUT + col0 + nn];
    }
    __syncthreads();

    const int w    = tid >> 5;            // wave 0..7
    const int lane = tid & 31;
    const int m0   = (w & 3) * 16;        // 4 row tiles
    const int n0   = (w >> 2) * 32;       // 2 col-tile pairs: {n0, n0+16}
    const int hm   = lane >> 4;           // K-half selector (ISA A/B layout)
    const int mr   = lane & 15;           // A row within tile / B column

    // A fragment: lane holds contiguous K = {k0+hm*16 .. +15} of its row
    const _Float16* apx = &As[(m0 + mr) * 256] + hm * 16;
    // B fragment: lane holds contiguous K = {k0+hm*16 .. +15} of its column
    const _Float16* bp0 = &Bs[(n0 + mr) * 256] + hm * 16;
    const _Float16* bp1 = &Bs[(n0 + 16 + mr) * 256] + hm * 16;

    v8f acc0 = {};
    v8f acc1 = {};
#pragma unroll
    for (int s = 0; s < 8; ++s) {
        const int k0 = s * 32;
        h8 a0 = *(const h8*)(apx + k0);
        h8 a1 = *(const h8*)(apx + k0 + 8);
        v16h af = __builtin_shufflevector(a0, a1,
                    0,1,2,3,4,5,6,7,8,9,10,11,12,13,14,15);
        v16h bf0 = *(const v16h*)(bp0 + k0);
        v16h bf1 = *(const v16h*)(bp1 + k0);
        acc0 = __builtin_amdgcn_wmma_f32_16x16x32_f16(
                   false, af, false, bf0, (short)0, acc0, false, false);
        acc1 = __builtin_amdgcn_wmma_f32_16x16x32_f16(
                   false, af, false, bf1, (short)0, acc1, false, false);
    }

    // C layout: vgpr r -> M = r + hm*8, N = mr
    const int gcol0 = col0 + n0 + mr;
    const float bb0 = bias[gcol0];
    const float bb1 = bias[gcol0 + 16];
    float* cp = C + (size_t)(row0 + m0 + hm * 8) * NOUT + gcol0;
#pragma unroll
    for (int r = 0; r < 8; ++r) {
        cp[r * NOUT]      = acc0[r] + bb0;
        cp[r * NOUT + 16] = acc1[r] + bb1;
    }
}

// ---------------------------------------------------------------------------
// Mask + sampling locations + softmax attention weights.
// One thread per (n, q, head). Rewrites sproj -> loc and aproj -> softmax(aw)
// in place (each element owned by exactly one thread).
// ---------------------------------------------------------------------------
__global__ __launch_bounds__(256) void mask_loc_aw_kernel(
    const float* __restrict__ mask_num,   // (N, 129, LQ, 1)
    const float* __restrict__ disp,       // scalar
    const float* __restrict__ refpts,     // (N, LQ, 4, 2)
    float* __restrict__ sproj,            // (N*LQ, 256) in: off proj, out: loc
    float* __restrict__ aproj)            // (N*LQ, 128) in: aw proj, out: softmax
{
    const int g = blockIdx.x * blockDim.x + threadIdx.x;
    if (g >= NB * LQN * MH) return;
    const int m  = g % MH;
    const int nq = g / MH;
    const int q  = nq % LQN;
    const int n  = nq / LQN;

    const float dis = disp[0];
    const float thr = mask_num[(n * 129 + 128) * LQN + q];
    const int sb = nq * 256;
    const int ab = nq * 128;

    const float invW[4] = { 1.0f/64.0f, 1.0f/32.0f, 1.0f/16.0f, 1.0f/8.0f };
    const float invH[4] = { 1.0f/64.0f, 1.0f/32.0f, 1.0f/16.0f, 1.0f/8.0f };

    float a[16];
    float amax = -1e30f;
#pragma unroll
    for (int t = 0; t < 16; ++t) {
        const int j = m * 16 + t;            // j = ((m*L)+l)*P + p
        const int l = t >> 2;
        const float mk = (mask_num[(n * 129 + j) * LQN + q] >= thr) ? 1.0f : 0.0f;
        const float ox = sproj[sb + 2 * j]     * mk;
        const float oy = sproj[sb + 2 * j + 1] * mk;
        const float rx = refpts[(nq * 4 + l) * 2];
        const float ry = refpts[(nq * 4 + l) * 2 + 1];
        sproj[sb + 2 * j]     = rx + ox * (dis * invW[l]);
        sproj[sb + 2 * j + 1] = ry + oy * (dis * invH[l]);
        const float av = aproj[ab + j] * mk;
        a[t] = av;
        amax = fmaxf(amax, av);
    }
    float sum = 0.0f;
#pragma unroll
    for (int t = 0; t < 16; ++t) { a[t] = __expf(a[t] - amax); sum += a[t]; }
    const float inv = 1.0f / sum;
#pragma unroll
    for (int t = 0; t < 16; ++t) aproj[ab + m * 16 + t] = a[t] * inv;
}

// ---------------------------------------------------------------------------
// Bilinear sampling + weighted reduce. One wave per (n, q, head); lane = DH
// channel (coalesced 128B corner loads). Lanes 0..15 hold the 16 (l,p)
// locations/weights, broadcast to the wave via __shfl.
// ---------------------------------------------------------------------------
__global__ __launch_bounds__(256) void sample_kernel(
    const float* __restrict__ value,      // (N, LEN_IN, M, DH)
    const float* __restrict__ loc,        // (N*LQ, 128, 2)
    const float* __restrict__ aw,         // (N*LQ, 128)
    float* __restrict__ mid)              // (N*LQ, 256)
{
    const int wave = blockIdx.x * 8 + (threadIdx.x >> 5);
    const int lane = threadIdx.x & 31;
    if (wave >= NB * LQN * MH) return;
    const int m  = wave % MH;
    const int nq = wave / MH;
    const int n  = nq / LQN;

    float lx = 0.0f, ly = 0.0f, wv = 0.0f;
    if (lane < 16) {
        const int j = m * 16 + lane;
        lx = loc[(size_t)nq * 256 + 2 * j];
        ly = loc[(size_t)nq * 256 + 2 * j + 1];
        wv = aw[(size_t)nq * 128 + j];
    }

    const int Hs[4]   = { 64, 32, 16, 8 };
    const int Wl[4]   = { 64, 32, 16, 8 };
    const int base[4] = { 0, 4096, 5120, 5376 };

    float acc = 0.0f;
#pragma unroll
    for (int lp = 0; lp < 16; ++lp) {
        const float sx = __shfl(lx, lp, 32);
        const float sy = __shfl(ly, lp, 32);
        const float w  = __shfl(wv, lp, 32);
        const int l  = lp >> 2;
        const int H  = Hs[l], W = Wl[l], vb = base[l];

        const float fx = sx * (float)W - 0.5f;
        const float fy = sy * (float)H - 0.5f;
        const float x0f = floorf(fx), y0f = floorf(fy);
        const float tx = fx - x0f, ty = fy - y0f;
        const int x0 = (int)x0f, y0 = (int)y0f;
#pragma unroll
        for (int c = 0; c < 4; ++c) {
            const int xi = x0 + (c & 1);
            const int yi = y0 + (c >> 1);
            const float cw = ((c & 1) ? tx : 1.0f - tx) * ((c >> 1) ? ty : 1.0f - ty);
            const bool ok = (xi >= 0) && (xi < W) && (yi >= 0) && (yi < H);
            const int xc = min(max(xi, 0), W - 1);
            const int yc = min(max(yi, 0), H - 1);
            const float g =
                value[(((size_t)n * LEN_IN + vb + yc * W + xc) * MH + m) * DH + lane];
            acc += ok ? g * (w * cw) : 0.0f;
        }
    }
    mid[(size_t)nq * 256 + m * DH + lane] = acc;
}

// ---------------------------------------------------------------------------
extern "C" void kernel_launch(void* const* d_in, const int* in_sizes, int n_in,
                              void* d_out, int out_size, void* d_ws, size_t ws_size,
                              hipStream_t stream) {
    const float* mask_num = (const float*)d_in[0];
    const float* dis      = (const float*)d_in[1];
    const float* query    = (const float*)d_in[2];
    const float* refpts   = (const float*)d_in[3];
    const float* inflat   = (const float*)d_in[4];
    // d_in[5] spatial shapes (int64, static), d_in[6] level_start_index: unused
    const float* Wv = (const float*)d_in[7];
    const float* bv = (const float*)d_in[8];
    const float* Ws = (const float*)d_in[9];
    const float* bs = (const float*)d_in[10];
    const float* Wa = (const float*)d_in[11];
    const float* ba = (const float*)d_in[12];
    const float* Wo = (const float*)d_in[13];
    const float* bo = (const float*)d_in[14];
    float* out = (float*)d_out;

    float* wsf   = (float*)d_ws;
    float* value = wsf;                                   // 21760*256
    float* sproj = value + (size_t)NQ_TOT * 256;          // 21760*256 (-> loc)
    float* aproj = sproj + (size_t)NQ_TOT * 256;          // 21760*128 (-> aw)
    float* mid   = aproj + (size_t)NQ_TOT * 128;          // 21760*256

    const dim3 blk(256);
    const int rowBlocks = NQ_TOT / 64;                    // 340

    // 1) value = input_flatten @ Wv + bv
    gemm256_wmma<256><<<dim3(rowBlocks, 4), blk, 0, stream>>>(inflat, Wv, bv, value);
    // 2) off proj = query @ Ws + bs
    gemm256_wmma<256><<<dim3(rowBlocks, 4), blk, 0, stream>>>(query, Ws, bs, sproj);
    // 3) aw proj = query @ Wa + ba
    gemm256_wmma<128><<<dim3(rowBlocks, 2), blk, 0, stream>>>(query, Wa, ba, aproj);
    // 4) mask, sampling locations, softmax (in place)
    mask_loc_aw_kernel<<<dim3((NB * LQN * MH + 255) / 256), blk, 0, stream>>>(
        mask_num, dis, refpts, sproj, aproj);
    // 5) bilinear sample + weighted sum -> mid
    sample_kernel<<<dim3(NB * LQN * MH / 8), blk, 0, stream>>>(value, sproj, aproj, mid);
    // 6) out = mid @ Wo + bo
    gemm256_wmma<256><<<dim3(rowBlocks, 4), blk, 0, stream>>>(mid, Wo, bo, out);
}
